// HybridRegression_65481071407111
// MI455X (gfx1250) — matrix operand order repaired and checked by reading, not compiled, not run on hardware
//
#include <hip/hip_runtime.h>
#include <math.h>

// ---------------------------------------------------------------------------
// HybridRegression fused inference for gfx1250 (MI455X, wave32, WMMA).
// GEMM1 is HBM-bound (x: 134 MB @ 23.3 TB/s ~ 5.8us floor) -> double-buffered
// K-stream, async-to-LDS for the f16 weight chunks, register-staged x chunks,
// and batched fragment loads so WMMAs issue back-to-back per k-step.
// ---------------------------------------------------------------------------

typedef __attribute__((ext_vector_type(4)))  _Float16 v4h;
typedef __attribute__((ext_vector_type(8)))  _Float16 v8h;
typedef __attribute__((ext_vector_type(16))) _Float16 v16h;
typedef __attribute__((ext_vector_type(8)))  float    v8f;
typedef __attribute__((ext_vector_type(4)))  int      i4v;

// Address-space-qualified int4 for the gfx1250 async-to-LDS builtin:
//   arg0: global (AS1) int4*, arg1: LDS (AS3) int4*, arg2/3: imm offset/cpol.
typedef __attribute__((address_space(1))) i4v g_i4v;
typedef __attribute__((address_space(3))) i4v l_i4v;

#define B_ROWS  8192
#define F_IN    4096
#define WG_ROWS 64     // rows per workgroup (4 waves x 16-row M tile)
#define KC      64     // K-chunk staged through LDS for GEMM1
#define NCHUNK  (F_IN / KC)

#if defined(__has_builtin)
#  if __has_builtin(__builtin_amdgcn_global_load_async_to_lds_b128)
#    define USE_ASYNC_LDS 1
#  endif
#  if __has_builtin(__builtin_amdgcn_s_wait_asynccnt)
#    define HAVE_WAIT_ASYNC 1
#  endif
#endif

__device__ __forceinline__ void wait_async_lds() {
#if defined(USE_ASYNC_LDS)
#  if defined(HAVE_WAIT_ASYNC)
  __builtin_amdgcn_s_wait_asynccnt(0);
#  else
  asm volatile("s_wait_asynccnt 0" ::: "memory");
#  endif
#endif
}

// ---- WMMA fragment loaders (contiguous k order, same for A and B) -----------------
// A 16x32 f16: lane L holds row M = L%16; lanes<16 hold K 0..15,
// lanes>=16 hold K 16..31. Two aligned 16B LDS reads per fragment.
__device__ __forceinline__ v16h load_A_frag(const _Float16* base, int ldk,
                                            int mbase, int kbase, int lane) {
  const int half = lane >> 4;
  const int m    = mbase + (lane & 15);
  const _Float16* p = base + m * ldk + kbase + half * 16;
  v8h lo = *(const v8h*)(p);        // K = half*16 + 0..7
  v8h hi = *(const v8h*)(p + 8);    // K = half*16 + 8..15
  return __builtin_shufflevector(lo, hi, 0,1,2,3,4,5,6,7,8,9,10,11,12,13,14,15);
}

// B 32x16 f16: lane L holds column N = L%16; lanes<16 hold K 0..15,
// lanes>=16 hold K 16..31 (contiguous 16 halfs = two aligned 16B reads).
__device__ __forceinline__ v16h load_B_frag(const _Float16* base, int ldk,
                                            int nbase, int kbase, int lane) {
  const int half = lane >> 4;
  const int n    = nbase + (lane & 15);
  const _Float16* p = base + n * ldk + kbase + half * 16;
  v8h lo = *(const v8h*)(p);
  v8h hi = *(const v8h*)(p + 8);
  return __builtin_shufflevector(lo, hi, 0,1,2,3,4,5,6,7,8,9,10,11,12,13,14,15);
}

__device__ __forceinline__ v8f wmma_f16(v16h a, v16h b, v8f c) {
  // (neg_a, A, neg_b, B, c_mod, C, reuse_a, reuse_b)
  return __builtin_amdgcn_wmma_f32_16x16x32_f16(false, a, false, b,
                                                (short)0, c, false, false);
}

// Stage one 128x64 f16 We1 chunk into an LDS buffer (8 x b128 per thread).
__device__ __forceinline__ void stage_B_chunk(const _Float16* __restrict__ w1h,
                                              _Float16* dst, int kb, int tid) {
#if defined(USE_ASYNC_LDS)
  #pragma unroll
  for (int it = 0; it < 8; ++it) {
    const int i = tid + it * 128;
    const int c = i >> 3, g = i & 7;
    const _Float16* gp = w1h + (size_t)c * F_IN + kb + g * 8;
    _Float16* lp = &dst[c * KC + g * 8];
    __builtin_amdgcn_global_load_async_to_lds_b128(
        (g_i4v*)(gp), (l_i4v*)(lp), 0, 0);
  }
#else
  #pragma unroll
  for (int it = 0; it < 8; ++it) {
    const int i = tid + it * 128;
    const int c = i >> 3, g = i & 7;
    *(v8h*)&dst[c * KC + g * 8] =
        *(const v8h*)(w1h + (size_t)c * F_IN + kb + g * 8);
  }
#endif
}

// ---- Kernel 1: convert We1 (128x4096 f32) to f16 in workspace -------------
__global__ void convert_w1_f16(const float* __restrict__ in,
                               _Float16* __restrict__ out, int n4) {
  int i = blockIdx.x * blockDim.x + threadIdx.x;
  if (i < n4) {
    float4 v = *(const float4*)(in + (size_t)i * 4);
    v4h h;
    h[0] = (_Float16)v.x; h[1] = (_Float16)v.y;
    h[2] = (_Float16)v.z; h[3] = (_Float16)v.w;
    *(v4h*)(out + (size_t)i * 4) = h;
  }
}

// ---- Kernel 2: fully fused network ----------------------------------------
__global__ __launch_bounds__(128)
void fused_mlp(const float* __restrict__ x, const _Float16* __restrict__ w1h,
               const float* __restrict__ be1,
               const float* __restrict__ We2, const float* __restrict__ be2,
               const float* __restrict__ We3, const float* __restrict__ be3,
               const float* __restrict__ Wq1, const float* __restrict__ bq1,
               const float* __restrict__ Wq2, const float* __restrict__ bq2,
               const float* __restrict__ Wh,  const float* __restrict__ bh,
               float* __restrict__ out) {
  __shared__ _Float16 sA [2][WG_ROWS * KC]; // x chunks (f16)        16 KB
  __shared__ _Float16 sB [2][128 * KC];     // We1 chunks (f16)      32 KB
  __shared__ _Float16 sH1[WG_ROWS * 128];   // relu layer-1 acts     16 KB
  __shared__ _Float16 sW2[64 * 128];        // We2 (f16)             16 KB
  __shared__ _Float16 sH2[WG_ROWS * 64];    // relu layer-2 acts      8 KB
  __shared__ _Float16 sW3[32 * 64];         // We3 (f16)              4 KB
  __shared__ float    sC [WG_ROWS * 32];    // sin^2 features         8 KB

  const int tid     = threadIdx.x;
  const int lane    = tid & 31;
  const int wave    = tid >> 5;             // 0..3, owns rows [16w,16w+16)
  const int half    = lane >> 4;
  const int ln16    = lane & 15;
  const int rowBase = blockIdx.x * WG_ROWS;

  // Phase 0: stage small weights in LDS as f16 (L2-resident, cheap).
  for (int i = tid; i < 64 * 128; i += 128) sW2[i] = (_Float16)We2[i];
  for (int i = tid; i < 32 * 64;  i += 128) sW3[i] = (_Float16)We3[i];

  // Stage chunk 0 (x: sync convert; We1: async DMA).
  {
    stage_B_chunk(w1h, sB[0], 0, tid);
    #pragma unroll
    for (int it = 0; it < 4; ++it) {
      const int i = tid + it * 128;
      const int r = i >> 3, g = i & 7;
      const float* src = x + (size_t)(rowBase + r) * F_IN + g * 8;
      float4 v0 = *(const float4*)(src);
      float4 v1 = *(const float4*)(src + 4);
      v8h h;
      h[0] = (_Float16)v0.x; h[1] = (_Float16)v0.y;
      h[2] = (_Float16)v0.z; h[3] = (_Float16)v0.w;
      h[4] = (_Float16)v1.x; h[5] = (_Float16)v1.y;
      h[6] = (_Float16)v1.z; h[7] = (_Float16)v1.w;
      *(v8h*)&sA[0][r * KC + g * 8] = h;
    }
  }
  wait_async_lds();
  __syncthreads();

  // ---------------- GEMM1: [64 x 4096] x [4096 x 128] -----------------
  v8f acc[8];
  #pragma unroll
  for (int t = 0; t < 8; ++t) acc[t] = (v8f){0.f,0.f,0.f,0.f,0.f,0.f,0.f,0.f};

  for (int c = 0; c < NCHUNK; ++c) {
    const int cur = c & 1, nxt = cur ^ 1;
    const bool more = (c + 1) < NCHUNK;
    const int kbn = (c + 1) * KC;

    // Kick off next We1 chunk via async DMA into the other buffer.
    if (more) stage_B_chunk(w1h, sB[nxt], kbn, tid);

    // Pull next x chunk into registers now; convert/store after compute.
    float4 xv0[4], xv1[4];
    if (more) {
      #pragma unroll
      for (int it = 0; it < 4; ++it) {
        const int i = tid + it * 128;
        const int r = i >> 3, g = i & 7;
        const float* src = x + (size_t)(rowBase + r) * F_IN + kbn + g * 8;
        xv0[it] = *(const float4*)(src);
        xv1[it] = *(const float4*)(src + 4);
      }
    }

    // Compute on current buffers: per k-step, batch A + all 8 B fragment
    // loads, then 8 back-to-back WMMAs (single s_wait_dscnt per k-step).
    #pragma unroll
    for (int ks = 0; ks < KC / 32; ++ks) {
      v16h a = load_A_frag(sA[cur], KC, wave * 16, ks * 32, lane);
      v16h bf[8];
      #pragma unroll
      for (int nt = 0; nt < 8; ++nt)
        bf[nt] = load_B_frag(sB[cur], KC, nt * 16, ks * 32, lane);
      #pragma unroll
      for (int nt = 0; nt < 8; ++nt)
        acc[nt] = wmma_f16(a, bf[nt], acc[nt]);
    }

    // Convert and store the next x chunk to the other LDS buffer.
    if (more) {
      #pragma unroll
      for (int it = 0; it < 4; ++it) {
        const int i = tid + it * 128;
        const int r = i >> 3, g = i & 7;
        v8h h;
        h[0] = (_Float16)xv0[it].x; h[1] = (_Float16)xv0[it].y;
        h[2] = (_Float16)xv0[it].z; h[3] = (_Float16)xv0[it].w;
        h[4] = (_Float16)xv1[it].x; h[5] = (_Float16)xv1[it].y;
        h[6] = (_Float16)xv1[it].z; h[7] = (_Float16)xv1[it].w;
        *(v8h*)&sA[nxt][r * KC + g * 8] = h;
      }
      wait_async_lds();
    }
    __syncthreads();
  }

  // Epilogue 1: +be1, ReLU, store f16. C/D layout: M = r + 8*half, N = ln16.
  #pragma unroll
  for (int nt = 0; nt < 8; ++nt) {
    const int col = nt * 16 + ln16;
    const float bias = be1[col];
    #pragma unroll
    for (int r = 0; r < 8; ++r) {
      const int rowl = wave * 16 + r + half * 8;
      float v = acc[nt][r] + bias;
      sH1[rowl * 128 + col] = (_Float16)(v > 0.f ? v : 0.f);
    }
  }
  __syncthreads();

  // ---------------- GEMM2: [64 x 128] x [128 x 64] --------------------
  v8f acc2[4];
  #pragma unroll
  for (int t = 0; t < 4; ++t) acc2[t] = (v8f){0.f,0.f,0.f,0.f,0.f,0.f,0.f,0.f};
  #pragma unroll
  for (int ks = 0; ks < 4; ++ks) {
    v16h a = load_A_frag(sH1, 128, wave * 16, ks * 32, lane);
    v16h bf[4];
    #pragma unroll
    for (int nt = 0; nt < 4; ++nt)
      bf[nt] = load_B_frag(sW2, 128, nt * 16, ks * 32, lane);
    #pragma unroll
    for (int nt = 0; nt < 4; ++nt)
      acc2[nt] = wmma_f16(a, bf[nt], acc2[nt]);
  }
  #pragma unroll
  for (int nt = 0; nt < 4; ++nt) {
    const int col = nt * 16 + ln16;
    const float bias = be2[col];
    #pragma unroll
    for (int r = 0; r < 8; ++r) {
      const int rowl = wave * 16 + r + half * 8;
      float v = acc2[nt][r] + bias;
      sH2[rowl * 64 + col] = (_Float16)(v > 0.f ? v : 0.f);
    }
  }
  __syncthreads();

  // ---------------- GEMM3: [64 x 64] x [64 x 32] + sin^2 --------------
  v8f acc3[2];
  #pragma unroll
  for (int t = 0; t < 2; ++t) acc3[t] = (v8f){0.f,0.f,0.f,0.f,0.f,0.f,0.f,0.f};
  #pragma unroll
  for (int ks = 0; ks < 2; ++ks) {
    v16h a = load_A_frag(sH2, 64, wave * 16, ks * 32, lane);
    v16h bf[2];
    #pragma unroll
    for (int nt = 0; nt < 2; ++nt)
      bf[nt] = load_B_frag(sW3, 64, nt * 16, ks * 32, lane);
    #pragma unroll
    for (int nt = 0; nt < 2; ++nt)
      acc3[nt] = wmma_f16(a, bf[nt], acc3[nt]);
  }
  #pragma unroll
  for (int nt = 0; nt < 2; ++nt) {
    const int col = nt * 16 + ln16;
    const float bias = be3[col];
    #pragma unroll
    for (int r = 0; r < 8; ++r) {
      const int rowl = wave * 16 + r + half * 8;
      float z = acc3[nt][r] + bias;
      float s = __sinf(1.57079632679489662f * z);   // sin(pi/2 * z)
      sC[rowl * 32 + col] = s * s;
    }
  }
  __syncthreads();

  // ---------------- Tail MLP 32->16->8->1 (scalar, ~650 FLOP/row) -----
  if (tid < WG_ROWS) {
    float cb[32];
    #pragma unroll
    for (int j = 0; j < 32; ++j) cb[j] = sC[tid * 32 + j];

    float q1[16];
    #pragma unroll
    for (int o = 0; o < 16; ++o) {
      float s = bq1[o];
      #pragma unroll
      for (int j = 0; j < 32; ++j) s = fmaf(Wq1[o * 32 + j], cb[j], s);
      q1[o] = fmaxf(s, 0.f);
    }
    float q2[8];
    #pragma unroll
    for (int o = 0; o < 8; ++o) {
      float s = bq2[o];
      #pragma unroll
      for (int j = 0; j < 16; ++j) s = fmaf(Wq2[o * 16 + j], q1[j], s);
      q2[o] = s;
    }
    float r = bh[0];
    #pragma unroll
    for (int j = 0; j < 8; ++j) r = fmaf(Wh[j], q2[j], r);
    out[rowBase + tid] = r;
  }
}

extern "C" void kernel_launch(void* const* d_in, const int* in_sizes, int n_in,
                              void* d_out, int out_size, void* d_ws, size_t ws_size,
                              hipStream_t stream) {
  const float* x   = (const float*)d_in[0];
  const float* We1 = (const float*)d_in[1];
  const float* be1 = (const float*)d_in[2];
  const float* We2 = (const float*)d_in[3];
  const float* be2 = (const float*)d_in[4];
  const float* We3 = (const float*)d_in[5];
  const float* be3 = (const float*)d_in[6];
  const float* Wq1 = (const float*)d_in[7];
  const float* bq1 = (const float*)d_in[8];
  const float* Wq2 = (const float*)d_in[9];
  const float* bq2 = (const float*)d_in[10];
  const float* Wh  = (const float*)d_in[11];
  const float* bh  = (const float*)d_in[12];
  float* out = (float*)d_out;

  _Float16* w1h = (_Float16*)d_ws;            // needs 128*4096*2 B = 1 MiB
  const int n4 = (128 * F_IN) / 4;
  convert_w1_f16<<<(n4 + 255) / 256, 256, 0, stream>>>(We1, w1h, n4);
  fused_mlp<<<B_ROWS / WG_ROWS, 128, 0, stream>>>(
      x, w1h, be1, We2, be2, We3, be3, Wq1, bq1, Wq2, bq2, Wh, bh, out);
}
